// SelfAttention_13640816132635
// MI455X (gfx1250) — matrix operand order repaired and checked by reading, not compiled, not run on hardware
//
#include <hip/hip_runtime.h>

// ===========================================================================
// SAGAN self-attention for MI455X (gfx1250, wave32).
// Round 3: ALL four matmuls on v_wmma_f32_16x16x32_f16. x is transposed once
// to f16 [n][l][256] (A-fragment-ready); 1x1 convs become WMMA GEMMs with the
// 2x2 maxpool performed in-register on C-layout accumulators; attention stays
// fused flash-style. TDM async tensor loads used for LDS staging (guarded).
// ===========================================================================

typedef __attribute__((ext_vector_type(16))) _Float16 v16h;
typedef __attribute__((ext_vector_type(8)))  float    v8f;

#define SN_EPS 1e-12f

// ---------------- workspace layout (bytes) ----------------
static constexpr size_t OFF_SIG   = 0;                                 // 4 floats
static constexpr size_t OFF_WTF   = 256;                               // 8kc*2ct*512 f16 = 16 KB
static constexpr size_t OFF_WPF   = OFF_WTF  + (size_t)8192*2;         // 16 KB
static constexpr size_t OFF_WGF   = OFF_WPF  + (size_t)8192*2;         // 8kc*8ct*512 = 64 KB
static constexpr size_t OFF_WOF   = OFF_WGF  + (size_t)32768*2;        // 4kc*16ct*512 = 64 KB
static constexpr size_t OFF_XT    = OFF_WOF  + (size_t)32768*2;        // [N][4096][256] f16 = 32 MB
static constexpr size_t OFF_THETA = OFF_XT   + (size_t)16*4096*256*2;  // [N][4096][32] f16
static constexpr size_t OFF_PHIF  = OFF_THETA+ (size_t)16*4096*32*2;   // [N][64][32][16] f16
static constexpr size_t OFF_GF    = OFF_PHIF + (size_t)16*64*512*2;    // [N][8][4][8][32][16] f16
static constexpr size_t OFF_ATTNG = OFF_GF   + (size_t)16*8*16384*2;   // [N][4096][128] f16

// ---------------- WMMA fragment helpers ----------------
union AFrag { v16h v; uint4 u[2]; };
union H8    { uint4 u; _Float16 h[8]; };

// A-matrix (16xK f16, K-chunk of 32) from row-major f16 (rowStride in halves).
__device__ __forceinline__ v16h load_A_f16(const _Float16* base, int rowStride, int k0) {
  const int lane = threadIdx.x & 31;
  const int row  = lane & 15;
  const int off  = (lane < 16) ? 0 : 16;
  const _Float16* p = base + (size_t)row * rowStride + k0 + off;
  AFrag a;
  a.u[0] = *reinterpret_cast<const uint4*>(p);
  a.u[1] = *reinterpret_cast<const uint4*>(p + 8);
  return a.v;
}

// B fragment from fragment-ready storage (two b128 loads per lane).
__device__ __forceinline__ v16h load_B_frag(const _Float16* frag) {
  const int lane = threadIdx.x & 31;
  const _Float16* p = frag + lane * 16;
  AFrag a;
  a.u[0] = *reinterpret_cast<const uint4*>(p);
  a.u[1] = *reinterpret_cast<const uint4*>(p + 8);
  return a.v;
}

__device__ __forceinline__ v8f wmma_f16(v16h a, v16h b, v8f c) {
  return __builtin_amdgcn_wmma_f32_16x16x32_f16(false, a, false, b, (short)0, c,
                                                false, false);
}

// ---------------- TDM async tensor load (guarded) ----------------
#if __has_builtin(__builtin_amdgcn_tensor_load_to_lds) && \
    __has_builtin(__builtin_amdgcn_s_wait_tensorcnt)
#define USE_TDM 1
#else
#define USE_TDM 0
#warning "CDNA5 TDM builtins not available on this toolchain; using synchronous LDS staging"
#endif

#if USE_TDM
typedef __attribute__((ext_vector_type(4))) unsigned int tdm_v4u;
typedef __attribute__((ext_vector_type(8))) int          tdm_v8i;
typedef __attribute__((ext_vector_type(4))) int          tdm_v4i;

// Flat shared pointers carry the LDS byte offset in their low 32 bits
// (ISA 10.2).
__device__ __forceinline__ unsigned lds_byte_addr(const void* p) {
  return (unsigned)(size_t)p;
}

// 1-D contiguous tile of `halves` f16 elements (halves <= 65535).
__device__ __forceinline__ void tdm_load_1d(const void* gsrc, void* ldst,
                                            unsigned halves) {
  const unsigned long long ga = (unsigned long long)(size_t)gsrc;
  tdm_v4u g0;
  g0[0] = 1u;                                                // count=1 user D#
  g0[1] = lds_byte_addr(ldst);                               // lds_addr
  g0[2] = (unsigned)(ga & 0xffffffffu);                      // global_addr lo
  g0[3] = (unsigned)((ga >> 32) & 0x1ffffffu) | (2u << 30);  // addr hi, type=2
  tdm_v8i g1;
  g1[0] = 1 << 16;                                           // data_size=1 (2B)
  g1[1] = (int)((halves & 0xffffu) << 16);                   // tensor_dim0 lo16
  g1[2] = (int)((halves >> 16) & 0xffffu) | (1 << 16);       // dim0 hi, dim1=1
  g1[3] = (int)((halves & 0xffffu) << 16);                   // tile_dim0
  g1[4] = 1;                                                 // tile_dim1=1
  g1[5] = (int)halves;                                       // dim0_stride lo
  g1[6] = 0;
  g1[7] = 0;
  tdm_v4i g2; g2[0] = 1; g2[1] = 1; g2[2] = (int)halves; g2[3] = 0;
  tdm_v4i g3; g3[0] = (int)halves; g3[1] = 1 << 16; g3[2] = 0; g3[3] = 0;
#if __clang_major__ >= 23
  tdm_v8i g4;
#pragma unroll
  for (int i = 0; i < 8; ++i) g4[i] = 0;
  __builtin_amdgcn_tensor_load_to_lds(g0, g1, g2, g3, g4, 0);
#else
  __builtin_amdgcn_tensor_load_to_lds(g0, g1, g2, g3, 0);
#endif
}
#endif  // USE_TDM

// Stage `halves` f16 from global into LDS (TDM when available).
__device__ __forceinline__ void stage_to_lds(const _Float16* gsrc, _Float16* ldst,
                                             unsigned halves, int tid, int nthr) {
#if USE_TDM
  if ((tid >> 5) == 0) {
    tdm_load_1d(gsrc, ldst, halves);
    __builtin_amdgcn_s_wait_tensorcnt(0);
  }
#else
  uint4* d = reinterpret_cast<uint4*>(ldst);
  const uint4* s = reinterpret_cast<const uint4*>(gsrc);
  const int nvec = (int)(halves >> 3);
  for (int i = tid; i < nvec; i += nthr) d[i] = s[i];
#endif
}

// ---------------- spectral-norm sigma (one power step) ----------------
__global__ __launch_bounds__(256) void sigma_kernel(
    const float* __restrict__ Wt, const float* __restrict__ Wp,
    const float* __restrict__ Wg, const float* __restrict__ Wo,
    const float* __restrict__ ut, const float* __restrict__ up,
    const float* __restrict__ ug, const float* __restrict__ uo,
    float* __restrict__ sig)
{
  __shared__ float vbuf[256];
  __shared__ float red[256];
  const int b = blockIdx.x, tid = threadIdx.x;
  const float* W; const float* u; int O, C;
  if      (b == 0) { W = Wt; u = ut; O = 32;  C = 256; }
  else if (b == 1) { W = Wp; u = up; O = 32;  C = 256; }
  else if (b == 2) { W = Wg; u = ug; O = 128; C = 256; }
  else             { W = Wo; u = uo; O = 256; C = 128; }

  float t1 = 0.f;
  if (tid < C)
    for (int o = 0; o < O; ++o) t1 += W[(size_t)o*C + tid] * u[o];
  red[tid] = (tid < C) ? t1 * t1 : 0.f;
  __syncthreads();
  for (int s = 128; s > 0; s >>= 1) {
    if (tid < s) red[tid] += red[tid + s];
    __syncthreads();
  }
  const float nrm1 = sqrtf(red[0]);
  __syncthreads();
  if (tid < C) vbuf[tid] = t1 / (nrm1 + SN_EPS);
  __syncthreads();

  float t2 = 0.f;
  if (tid < O)
    for (int c = 0; c < C; ++c) t2 += W[(size_t)tid*C + c] * vbuf[c];
  red[tid] = (tid < O) ? t2 * t2 : 0.f;
  __syncthreads();
  for (int s = 128; s > 0; s >>= 1) {
    if (tid < s) red[tid] += red[tid + s];
    __syncthreads();
  }
  if (tid == 0) {
    const float s2 = red[0];
    const float sigma = s2 / (sqrtf(s2) + SN_EPS);
    sig[b] = 1.0f / sigma;
  }
}

// ---------------- weight -> B-fragment pack (generic) ----------------
// B element (K=cc in, N=co out) taken from W[co][cc]*invsig; layout
// frags[kc(Cin/32)][ct(Cout/16)][lane(32)][i(16)].
__global__ void pack_w_kernel(const float* __restrict__ W,
                              const float* __restrict__ sig, int sigIdx,
                              int Cin, int nCt, _Float16* __restrict__ frags)
{
  const int i = blockIdx.x * 256 + threadIdx.x;
  if (i >= Cin * nCt * 16) return;
  const int co = i / Cin, cc = i % Cin;
  const int kc = cc >> 5, q = cc & 31;
  const int ln = (co & 15) | ((q >> 4) << 4);
  const int ii = q & 15;
  const int ct = co >> 4;
  frags[(((size_t)kc*nCt + ct)*32 + ln)*16 + ii] =
      (_Float16)(W[(size_t)co*Cin + cc] * sig[sigIdx]);
}

// ---------------- x (f32 [n][c][l]) -> xT (f16 [n][l][256]) ----------------
// 64x64 LDS tile transpose; A-fragment-ready output.
__global__ __launch_bounds__(256) void xcvt_kernel(const float* __restrict__ x,
                                                   _Float16* __restrict__ xT)
{
  __shared__ _Float16 T[64][72];     // +8 pad vs bank conflicts
  const int tid = threadIdx.x;
  const int n   = blockIdx.y;
  const int lt  = blockIdx.x & 63;   // 64 l-tiles
  const int cb  = blockIdx.x >> 6;   // 4 c-tiles
  const int l0  = lt * 64, c0 = cb * 64;

#pragma unroll
  for (int i = 0; i < 16; ++i) {
    const int idx = i*256 + tid;               // 0..4095
    const int c = idx >> 6, l = idx & 63;
    T[c][l] = (_Float16)x[((size_t)n*256 + c0 + c)*4096 + l0 + l];
  }
  __syncthreads();
#pragma unroll
  for (int i = 0; i < 2; ++i) {
    const int task = i*256 + tid;              // 0..511 : (l, c-octet)
    const int l = task >> 3, oc = task & 7;
    H8 h;
#pragma unroll
    for (int j = 0; j < 8; ++j) h.h[j] = T[oc*8 + j][l];
    *reinterpret_cast<uint4*>(xT + ((size_t)n*4096 + l0 + l)*256 + c0 + oc*8) = h.u;
  }
}

// ---------------- WMMA 1x1 conv (+ in-register 2x2 maxpool) ---------------
// D(l x Cout) = xT(l x 256) . W^T(256 x Cout), K = 256 = 8 WMMA.
// mode 0: theta -> [n][l][32].  mode 1: phi -> phiF frags (pooled).
// mode 2: g -> gF frags (pooled).  Pool pairs rows (M,M+1) = adjacent C VGPRs
// in the same lane, and h-adjacent l-tiles t / t+4 handled by the same wave.
__global__ __launch_bounds__(256) void conv_kernel(
    const _Float16* __restrict__ xT, const _Float16* __restrict__ Wfrag,
    _Float16* __restrict__ out, int mode, int nCt)
{
  __shared__ __align__(16) _Float16 WsF[8*8*512];  // up to 64 KB of B-fragments
  const int tid = threadIdx.x, lane = tid & 31, wave = tid >> 5;
  const int n = blockIdx.y;
  const int lbase = blockIdx.x * 256;

  stage_to_lds(Wfrag, WsF, (unsigned)(nCt * 8 * 512), tid, 256);
  __syncthreads();

  // tile assignment: pooled -> h-adjacent pair (i1, i1+4); plain -> (w, w+8)
  const bool pool = (mode != 0);
  const int i1 = pool ? ((wave & 3) | ((wave & 4) << 1)) : wave;
  const int i2 = pool ? (i1 + 4) : (wave + 8);
  const int l1 = lbase + i1 * 16;
  const int l2 = lbase + i2 * 16;

  // preload A fragments for both l-tiles (K = 256 -> 8 chunks)
  v16h A1[8], A2[8];
  const _Float16* a1 = xT + ((size_t)n*4096 + l1) * 256;
  const _Float16* a2 = xT + ((size_t)n*4096 + l2) * 256;
#pragma unroll
  for (int kc = 0; kc < 8; ++kc) {
    A1[kc] = load_A_f16(a1, 256, kc*32);
    A2[kc] = load_A_f16(a2, 256, kc*32);
  }

  for (int ct = 0; ct < nCt; ++ct) {
    v8f c1 = {}, c2 = {};
#pragma unroll
    for (int kc = 0; kc < 8; ++kc) {
      const v16h B = load_B_frag(WsF + ((size_t)kc*nCt + ct)*512);
      c1 = wmma_f16(A1[kc], B, c1);
      c2 = wmma_f16(A2[kc], B, c2);
    }

    if (mode == 0) {                    // theta: [n][l][32] (A-ready rows)
#pragma unroll
      for (int r = 0; r < 8; ++r) {
        const int M = r + ((lane < 16) ? 0 : 8);
        const int o = ct*16 + (lane & 15);
        out[((size_t)n*4096 + l1 + M)*32 + o] = (_Float16)c1[r];
        out[((size_t)n*4096 + l2 + M)*32 + o] = (_Float16)c2[r];
      }
    } else {
      // 2x2 maxpool: rows (2j,2j+1) of tiles l1 (h) and l2=l1+64 (h+1)
      const int c = ct*16 + (lane & 15);
      const int wp0 = (l1 & 63) >> 1;
      const int hp  = l1 >> 7;
#pragma unroll
      for (int j = 0; j < 4; ++j) {
        const float pm = fmaxf(fmaxf(c1[2*j], c1[2*j+1]),
                               fmaxf(c2[2*j], c2[2*j+1]));
        const int wp = wp0 + j + ((lane < 16) ? 0 : 4);
        const int md = hp*32 + wp;
        if (mode == 1) {                // phiF: B elem (K=c<32, N=md)
          const int t  = md >> 4;
          const int ln = (md & 15) | ((c >> 4) << 4);
          const int ii = c & 15;
          out[(((size_t)n*64 + t)*32 + ln)*16 + ii] = (_Float16)pm;
        } else {                        // gF: B elem (K=md, N=c)
          const int mc = md >> 7, rr = md & 127;
          const int kc2 = rr >> 5, q = rr & 31;
          const int ln = (c & 15) | ((q >> 4) << 4);
          const int ii = q & 15;
          const int ct2 = (c >> 4) & 7;
          out[((((size_t)n*8 + mc)*4 + kc2)*8 + ct2)*512 + (size_t)ln*16 + ii] =
              (_Float16)pm;
        }
      }
    }
  }
}

// ---------------- fused attention: softmax(theta^T phi) @ g^T -------------
__global__ __launch_bounds__(128) void attn_kernel(
    const _Float16* __restrict__ theta, const _Float16* __restrict__ phiF,
    const _Float16* __restrict__ gF,    _Float16* __restrict__ attng)
{
  __shared__ __align__(16) _Float16 phiC[8*512];      //  8 KB: 8 B-fragments
  __shared__ __align__(16) _Float16 gC[32*512];       // 32 KB: 4kc x 8ct frags
  __shared__ __align__(16) _Float16 pStage[4*16*128]; // 16 KB per-wave [M][m]

  const int tid  = threadIdx.x;
  const int lane = tid & 31;
  const int wave = tid >> 5;
  const int n    = blockIdx.y;
  const int l0   = blockIdx.x * 64 + wave * 16;

  const _Float16* phiFn = phiF + (size_t)n * 64 * 512;
  const _Float16* gFn   = gF   + (size_t)n * 8  * 16384;

  const v16h Ath = load_A_f16(theta + ((size_t)n*4096 + l0)*32, 32, 0);

  const v8f vzero = {};
  v8f O[8];
#pragma unroll
  for (int i = 0; i < 8; ++i) O[i] = vzero;
  float rowM[8], rowL[8];
#pragma unroll
  for (int r = 0; r < 8; ++r) { rowM[r] = -3.4e38f; rowL[r] = 0.f; }

  _Float16* ps = pStage + wave * (16*128);

  for (int mc = 0; mc < 8; ++mc) {
    __syncthreads();   // previous chunk fully consumed
#if USE_TDM
    if (wave == 0)
      tdm_load_1d(gFn + (size_t)mc*16384, gC, 16384);
    {
      uint4* dp = reinterpret_cast<uint4*>(phiC);
      const uint4* sp = reinterpret_cast<const uint4*>(phiFn + (size_t)mc*4096);
#pragma unroll
      for (int i = 0; i < 4; ++i) dp[i*128 + tid] = sp[i*128 + tid];
    }
    if (wave == 0) __builtin_amdgcn_s_wait_tensorcnt(0);
#else
    {
      uint4* dp = reinterpret_cast<uint4*>(phiC);
      const uint4* sp = reinterpret_cast<const uint4*>(phiFn + (size_t)mc*4096);
#pragma unroll
      for (int i = 0; i < 4; ++i) dp[i*128 + tid] = sp[i*128 + tid];
      uint4* dg = reinterpret_cast<uint4*>(gC);
      const uint4* sg = reinterpret_cast<const uint4*>(gFn + (size_t)mc*16384);
#pragma unroll
      for (int i = 0; i < 16; ++i) dg[i*128 + tid] = sg[i*128 + tid];
    }
#endif
    __syncthreads();

    v8f S[8];
#pragma unroll
    for (int mt = 0; mt < 8; ++mt)
      S[mt] = wmma_f16(Ath, load_B_frag(phiC + mt*512), vzero);

#pragma unroll
    for (int r = 0; r < 8; ++r) {
      float mx = -3.4e38f;
#pragma unroll
      for (int mt = 0; mt < 8; ++mt) mx = fmaxf(mx, S[mt][r]);
#pragma unroll
      for (int d = 8; d >= 1; d >>= 1) mx = fmaxf(mx, __shfl_xor(mx, d, 32));
      const float mnew  = fmaxf(rowM[r], mx);
      const float alpha = __expf(rowM[r] - mnew);
      rowM[r] = mnew;
      float s = 0.f;
#pragma unroll
      for (int mt = 0; mt < 8; ++mt) {
        const float p = __expf(S[mt][r] - mnew);
        S[mt][r] = p;
        s += p;
      }
#pragma unroll
      for (int d = 8; d >= 1; d >>= 1) s += __shfl_xor(s, d, 32);
      rowL[r] = rowL[r] * alpha + s;
#pragma unroll
      for (int ct = 0; ct < 8; ++ct) O[ct][r] *= alpha;
    }

#pragma unroll
    for (int mt = 0; mt < 8; ++mt)
#pragma unroll
      for (int r = 0; r < 8; ++r) {
        const int M = r + ((lane < 16) ? 0 : 8);
        ps[M*128 + mt*16 + (lane & 15)] = (_Float16)S[mt][r];
      }

#pragma unroll
    for (int kc = 0; kc < 4; ++kc) {
      const v16h Ap = load_A_f16(ps, 128, kc*32);
#pragma unroll
      for (int ct = 0; ct < 8; ++ct)
        O[ct] = wmma_f16(Ap, load_B_frag(gC + (kc*8 + ct)*512), O[ct]);
    }
  }

#pragma unroll
  for (int r = 0; r < 8; ++r) {
    const float inv = 1.0f / rowL[r];
    const int l = l0 + r + ((lane < 16) ? 0 : 8);
    _Float16* dst = attng + ((size_t)n*4096 + l)*128 + (lane & 15);
#pragma unroll
    for (int ct = 0; ct < 8; ++ct)
      dst[ct*16] = (_Float16)(O[ct][r] * inv);
  }
}

// ---------------- output 1x1 conv + residual ------------------------------
__global__ __launch_bounds__(256) void outconv_kernel(
    const float* __restrict__ x, const _Float16* __restrict__ attng,
    const _Float16* __restrict__ WoF, const float* __restrict__ gamma_p,
    float* __restrict__ out)
{
  __shared__ __align__(16) _Float16 Bw[64*512];   // 64 KB: 4kc x 16ct frags
  const int tid = threadIdx.x, lane = tid & 31, wave = tid >> 5;
  const int n = blockIdx.y, l0 = blockIdx.x * 16;

  stage_to_lds(WoF, Bw, 32768u, tid, 256);
  __syncthreads();

  v16h A[4];
  const _Float16* abase = attng + ((size_t)n*4096 + l0)*128;
#pragma unroll
  for (int kc = 0; kc < 4; ++kc) A[kc] = load_A_f16(abase, 128, kc*32);

  const float gm = gamma_p[0];

#pragma unroll
  for (int t = 0; t < 2; ++t) {
    const int ct = wave + t*8;
    v8f acc = {};
#pragma unroll
    for (int kc = 0; kc < 4; ++kc)
      acc = wmma_f16(A[kc], load_B_frag(Bw + (kc*16 + ct)*512), acc);

    const int co = ct*16 + (lane & 15);
#pragma unroll
    for (int r = 0; r < 8; ++r) {
      const int l = l0 + r + ((lane < 16) ? 0 : 8);
      const size_t idx = ((size_t)n*256 + co)*4096 + l;
      out[idx] = x[idx] + gm * acc[r];
    }
  }
}

// ---------------- launch --------------------------------------------------
extern "C" void kernel_launch(void* const* d_in, const int* in_sizes, int n_in,
                              void* d_out, int out_size, void* d_ws, size_t ws_size,
                              hipStream_t stream) {
  (void)in_sizes; (void)n_in; (void)out_size; (void)ws_size;
  const float* x     = (const float*)d_in[0];
  const float* Wt    = (const float*)d_in[1];
  const float* Wp    = (const float*)d_in[2];
  const float* Wg    = (const float*)d_in[3];
  const float* Wo    = (const float*)d_in[4];
  const float* ut    = (const float*)d_in[5];
  const float* up    = (const float*)d_in[6];
  const float* ug    = (const float*)d_in[7];
  const float* uo    = (const float*)d_in[8];
  const float* gamma = (const float*)d_in[9];
  float* out = (float*)d_out;

  char* ws = (char*)d_ws;
  float*    sig   = (float*)   (ws + OFF_SIG);
  _Float16* WtF   = (_Float16*)(ws + OFF_WTF);
  _Float16* WpF   = (_Float16*)(ws + OFF_WPF);
  _Float16* WgF   = (_Float16*)(ws + OFF_WGF);
  _Float16* WoF   = (_Float16*)(ws + OFF_WOF);
  _Float16* xT    = (_Float16*)(ws + OFF_XT);
  _Float16* theta = (_Float16*)(ws + OFF_THETA);
  _Float16* phiF  = (_Float16*)(ws + OFF_PHIF);
  _Float16* gF    = (_Float16*)(ws + OFF_GF);
  _Float16* attng = (_Float16*)(ws + OFF_ATTNG);

  sigma_kernel<<<4, 256, 0, stream>>>(Wt, Wp, Wg, Wo, ut, up, ug, uo, sig);
  pack_w_kernel<<< 32, 256, 0, stream>>>(Wt, sig, 0, 256,  2, WtF);
  pack_w_kernel<<< 32, 256, 0, stream>>>(Wp, sig, 1, 256,  2, WpF);
  pack_w_kernel<<<128, 256, 0, stream>>>(Wg, sig, 2, 256,  8, WgF);
  pack_w_kernel<<<128, 256, 0, stream>>>(Wo, sig, 3, 128, 16, WoF);
  xcvt_kernel<<<dim3(256,16), 256, 0, stream>>>(x, xT);
  conv_kernel<<<dim3(16,16), 256, 0, stream>>>(xT, WtF, theta, 0, 2);
  conv_kernel<<<dim3(16,16), 256, 0, stream>>>(xT, WpF, phiF,  1, 2);
  conv_kernel<<<dim3(16,16), 256, 0, stream>>>(xT, WgF, gF,    2, 8);
  attn_kernel<<<dim3(64,16), 128, 0, stream>>>(theta, phiF, gF, attng);
  outconv_kernel<<<dim3(256,16), 256, 0, stream>>>(x, attng, WoF, gamma, out);
}
